// PNA_NET_19877108646249
// MI455X (gfx1250) — matrix operand order repaired and checked by reading, not compiled, not run on hardware
//
#include <hip/hip_runtime.h>
#include <hip/hip_bf16.h>

#define N_NODES 50000
#define N_EDGES 600000
#define DIM     128
#define LAYERS  4
#define GRAPHS  512
#define TOUT    10
#define EPS     1e-5f

typedef __bf16 bf16;
typedef bf16  v8bf  __attribute__((ext_vector_type(8)));
typedef bf16  v16bf __attribute__((ext_vector_type(16)));
typedef float v8f   __attribute__((ext_vector_type(8)));

// ---------- helpers ----------
__device__ __forceinline__ bf16 f2bf(float f) {
    unsigned u = __float_as_uint(f);
    unsigned r = u + 0x7FFFu + ((u >> 16) & 1u);   // RNE to bf16
    unsigned short h = (unsigned short)(r >> 16);
    bf16 out;
    __builtin_memcpy(&out, &h, 2);
    return out;
}
// monotonic float <-> uint encoding for integer atomicMin/Max
__device__ __forceinline__ unsigned fenc(float f) {
    unsigned u = __float_as_uint(f);
    return (u & 0x80000000u) ? ~u : (u | 0x80000000u);
}
__device__ __forceinline__ float fdec(unsigned u) {
    return __uint_as_float((u & 0x80000000u) ? (u & 0x7FFFFFFFu) : ~u);
}

// ---------- generic fills ----------
__global__ void k_fill_f32(float* p, float v, int n) {
    int i = blockIdx.x * blockDim.x + threadIdx.x;
    if (i < n) p[i] = v;
}

// ---------- degree / scalers ----------
__global__ void k_deg(const int* __restrict__ dst, float* __restrict__ deg, int e) {
    int i = blockIdx.x * blockDim.x + threadIdx.x;
    if (i < e) atomicAdd(&deg[dst[i]], 1.0f);
}
__global__ void k_logsum(const float* __restrict__ deg, float* __restrict__ sum, int n) {
    __shared__ float sh[256];
    int i = blockIdx.x * blockDim.x + threadIdx.x;
    float v = (i < n) ? log1pf(deg[i]) : 0.0f;
    sh[threadIdx.x] = v;
    __syncthreads();
    for (int s = 128; s > 0; s >>= 1) {
        if (threadIdx.x < s) sh[threadIdx.x] += sh[threadIdx.x + s];
        __syncthreads();
    }
    if (threadIdx.x == 0) atomicAdd(sum, sh[0]);
}
__global__ void k_ampatt(const float* __restrict__ deg, const float* __restrict__ sum,
                         float* __restrict__ amp, float* __restrict__ att, int n) {
    int i = blockIdx.x * blockDim.x + threadIdx.x;
    if (i >= n) return;
    float delta = *sum / (float)n;
    float d  = fmaxf(deg[i], 1.0f);
    float ld = log1pf(d);
    amp[i] = ld / delta;
    att[i] = delta / ld;
}

// ---------- node embedding: h = x @ emb_W + emb_b ----------
__global__ void k_embed(const float* __restrict__ x, const float* __restrict__ W,
                        const float* __restrict__ b, float* __restrict__ h, int total) {
    int i = blockIdx.x * blockDim.x + threadIdx.x;
    if (i >= total) return;
    int node = i >> 7, f = i & 127;
    const float* xr = x + node * 3;
    h[i] = xr[0] * W[f] + xr[1] * W[DIM + f] + xr[2] * W[2 * DIM + f] + b[f];
}

// ---------- weight convert: conv_W[l][k][n] (f32) -> Wbt[l][n][k] (bf16, K-contiguous) ----------
__global__ void k_convw(const float* __restrict__ W, bf16* __restrict__ Wbt, int total) {
    int i = blockIdx.x * blockDim.x + threadIdx.x;
    if (i >= total) return;
    int l   = i / (12 * DIM * DIM);
    int rem = i % (12 * DIM * DIM);
    int k   = rem / DIM;
    int n   = rem % DIM;
    Wbt[((size_t)l * DIM + n) * (12 * DIM) + k] = f2bf(W[i]);
}

// ---------- per-layer aggregation buffer init: [N][512] = s1 | min_enc | max_enc | s2 ----------
__global__ void k_init_agg(float* __restrict__ agg, int total) {
    int i = blockIdx.x * blockDim.x + threadIdx.x;
    if (i >= total) return;
    int r = i & 511;
    unsigned* u = (unsigned*)agg;
    if (r < 128)      agg[i] = 0.0f;         // s1
    else if (r < 256) u[i]   = 0xFFFFFFFFu;  // encoded +inf for min
    else if (r < 384) u[i]   = 0x00000000u;  // encoded -inf for max
    else              agg[i] = 0.0f;         // s2
}

// ---------- edge scatter: 4 segment reductions ----------
__global__ void k_scatter(const int* __restrict__ src, const int* __restrict__ dst,
                          const float* __restrict__ h, float* __restrict__ agg, int total) {
    int i = blockIdx.x * blockDim.x + threadIdx.x;
    if (i >= total) return;
    int e = i >> 7, f = i & 127;
    float m = h[(size_t)src[e] * DIM + f];
    size_t base = (size_t)dst[e] * 512 + f;
    atomicAdd(&agg[base], m);
    atomicAdd(&agg[base + 384], m * m);
    unsigned enc = fenc(m);
    atomicMin((unsigned*)agg + base + 128, enc);
    atomicMax((unsigned*)agg + base + 256, enc);
}

// ---------- finalize aggregations -> bf16 A matrix [N][512] = mean | min | max | std ----------
__global__ void k_fixup(const float* __restrict__ agg, const float* __restrict__ deg,
                        bf16* __restrict__ aggB, int total) {
    int i = blockIdx.x * blockDim.x + threadIdx.x;
    if (i >= total) return;
    int node = i >> 7, f = i & 127;
    float dg = deg[node];
    float d  = fmaxf(dg, 1.0f);
    size_t base = (size_t)node * 512 + f;
    float s1 = agg[base], s2 = agg[base + 384];
    float mean = s1 / d;
    float var  = fmaxf(s2 / d - mean * mean, 0.0f);
    float sd   = sqrtf(var + EPS);
    bool he = dg > 0.0f;
    float mn = he ? fdec(((const unsigned*)agg)[base + 128]) : 0.0f;
    float mx = he ? fdec(((const unsigned*)agg)[base + 256]) : 0.0f;
    aggB[base]       = f2bf(mean);
    aggB[base + 128] = f2bf(mn);
    aggB[base + 256] = f2bf(mx);
    aggB[base + 384] = f2bf(sd);
}

// ---------- WMMA GEMM: out = A@W0 + amp*(A@W1) + att*(A@W2) + bias ----------
// Workgroup = 4 waves = 4 row tiles sharing one column tile; B panel staged in LDS.
#define BROWT   4
#define BSTRIDE 1544   // padded row stride in halves: 772 dwords == 4 mod 64 banks

__device__ __forceinline__ v16bf ld_frag_g(const bf16* p0, const bf16* p1) {
    v8bf lo = *(const v8bf*)p0;
    v8bf hi = *(const v8bf*)p1;
    return __builtin_shufflevector(lo, hi, 0, 1, 2, 3, 4, 5, 6, 7,
                                           8, 9, 10, 11, 12, 13, 14, 15);
}

__global__ __launch_bounds__(128)
void k_gemm(const bf16* __restrict__ aggB, const bf16* __restrict__ Wbt,
            const float* __restrict__ amp, const float* __restrict__ att,
            const float* __restrict__ bias, float* __restrict__ out) {
    __shared__ bf16 Bs[16 * BSTRIDE];        // 16 columns x 1536 K (padded)

    const int ROWT = N_NODES / 16;           // 3125 row tiles
    int wid  = threadIdx.x >> 5;
    int lane = threadIdx.x & 31;
    int ct   = blockIdx.x & 7;               // 8 column tiles
    int rg   = blockIdx.x >> 3;              // row-tile group of 4

    // cooperative B panel copy: source is a contiguous 16x1536 bf16 block
    {
        const bf16* srcp = Wbt + (size_t)ct * 16 * (12 * DIM);
        for (int idx = threadIdx.x; idx < 16 * 192; idx += 128) {   // 192 x 8-half chunks/row
            int rr = idx / 192, cc = idx % 192;
            *(v8bf*)(Bs + rr * BSTRIDE + cc * 8) = *(const v8bf*)(srcp + rr * 1536 + cc * 8);
        }
    }
    __syncthreads();

    int rt = rg * BROWT + wid;
    if (rt >= ROWT) return;                  // idle tail wave (after barrier)

    int row0 = rt * 16;
    int m    = lane & 15;
    int hi   = lane >> 4;

    // A: 16x32 bf16 fragment (lane row = m; K base 16*hi; halves at +0 and +8)
    const bf16* aRow = aggB + (size_t)(row0 + m) * 512 + hi * 16;
    // B: 32x16 bf16 fragment from LDS; K-contiguous per column (16 K per lane half)
    const bf16* bCol = Bs + m * BSTRIDE + hi * 16;

    v8f acc0 = {}, acc1 = {}, acc2 = {};
    for (int kk = 0; kk < 512; kk += 32) {
        v16bf a  = ld_frag_g(aRow + kk, aRow + kk + 8);
        v16bf b0 = ld_frag_g(bCol + kk,        bCol + kk + 8);
        v16bf b1 = ld_frag_g(bCol + 512 + kk,  bCol + 512 + kk + 8);
        v16bf b2 = ld_frag_g(bCol + 1024 + kk, bCol + 1024 + kk + 8);
        acc0 = __builtin_amdgcn_wmma_f32_16x16x32_bf16(false, a, false, b0, (short)0, acc0, false, false);
        acc1 = __builtin_amdgcn_wmma_f32_16x16x32_bf16(false, a, false, b1, (short)0, acc1, false, false);
        acc2 = __builtin_amdgcn_wmma_f32_16x16x32_bf16(false, a, false, b2, (short)0, acc2, false, false);
    }

    int col = ct * 16 + m;
    float bv = bias[col];
    #pragma unroll
    for (int r = 0; r < 8; ++r) {
        int row = row0 + r + hi * 8;         // C layout: M = r + 8*(lane>=16)
        float o = acc0[r] + amp[row] * acc1[r] + att[row] * acc2[r] + bv;
        out[(size_t)row * DIM + col] = o;
    }
}

// ---------- batchnorm stats (per-column sum / sumsq over N rows) ----------
__global__ void k_bnstat(const float* __restrict__ out, float* __restrict__ bnsum,
                         float* __restrict__ bnsq, int nrows) {
    int f = threadIdx.x;                     // 128 threads
    float s = 0.0f, q = 0.0f;
    for (int r = blockIdx.x; r < nrows; r += gridDim.x) {
        float v = out[(size_t)r * DIM + f];
        s += v; q += v * v;
    }
    atomicAdd(&bnsum[f], s);
    atomicAdd(&bnsq[f],  q);
}

// ---------- batchnorm apply + relu + residual into h ----------
__global__ void k_bnapply(const float* __restrict__ out, const float* __restrict__ bnsum,
                          const float* __restrict__ bnsq, const float* __restrict__ gamma,
                          const float* __restrict__ beta, float* __restrict__ h, int total) {
    int i = blockIdx.x * blockDim.x + threadIdx.x;
    if (i >= total) return;
    int f = i & 127;
    const float invN = 1.0f / (float)N_NODES;
    float mu  = bnsum[f] * invN;
    float var = bnsq[f] * invN - mu * mu;
    float o = (out[i] - mu) * rsqrtf(var + EPS) * gamma[f] + beta[f];
    h[i] += fmaxf(o, 0.0f);
}

// ---------- global mean pool ----------
__global__ void k_pool(const float* __restrict__ h, const int* __restrict__ batch,
                       float* __restrict__ gsum, float* __restrict__ gcnt, int total) {
    int i = blockIdx.x * blockDim.x + threadIdx.x;
    if (i >= total) return;
    int node = i >> 7, f = i & 127;
    int g = batch[node];
    atomicAdd(&gsum[(size_t)g * DIM + f], h[i]);
    if (f == 0) atomicAdd(&gcnt[g], 1.0f);
}
__global__ void k_gmean(float* __restrict__ gsum, const float* __restrict__ gcnt, int total) {
    int i = blockIdx.x * blockDim.x + threadIdx.x;
    if (i >= total) return;
    gsum[i] /= fmaxf(gcnt[i >> 7], 1.0f);
}

// ---------- tiny MLP tail ----------
__global__ void k_mlp1(const float* __restrict__ g, const float* __restrict__ W,
                       const float* __restrict__ b, float* __restrict__ t, int total) {
    int i = blockIdx.x * blockDim.x + threadIdx.x;
    if (i >= total) return;
    int r = i >> 6, c = i & 63;              // [512,128]@[128,64]
    float acc = b[c];
    const float* gr = g + (size_t)r * DIM;
    for (int k = 0; k < DIM; ++k) acc += gr[k] * W[k * 64 + c];
    t[i] = fmaxf(acc, 0.0f);
}
__global__ void k_mlp2(const float* __restrict__ t1, const float* __restrict__ W,
                       const float* __restrict__ b, float* __restrict__ t2, int total) {
    int i = blockIdx.x * blockDim.x + threadIdx.x;
    if (i >= total) return;
    int r = i >> 5, c = i & 31;              // [512,64]@[64,32]
    float acc = b[c];
    const float* tr = t1 + (size_t)r * 64;
    for (int k = 0; k < 64; ++k) acc += tr[k] * W[k * 32 + c];
    t2[i] = fmaxf(acc, 0.0f);
}
__global__ void k_mlp3(const float* __restrict__ t2, const float* __restrict__ W,
                       const float* __restrict__ b, float* __restrict__ out, int total) {
    int i = blockIdx.x * blockDim.x + threadIdx.x;
    if (i >= total) return;
    int r = i / TOUT, c = i % TOUT;          // [512,32]@[32,10]
    float acc = b[c];
    const float* tr = t2 + (size_t)r * 32;
    for (int k = 0; k < 32; ++k) acc += tr[k] * W[k * TOUT + c];
    out[i] = acc;
}

static inline int gdiv(long long a, int b) { return (int)((a + b - 1) / b); }

extern "C" void kernel_launch(void* const* d_in, const int* in_sizes, int n_in,
                              void* d_out, int out_size, void* d_ws, size_t ws_size,
                              hipStream_t stream) {
    const float* x      = (const float*)d_in[0];
    const int*   ei     = (const int*)d_in[1];
    const int*   batch  = (const int*)d_in[2];
    const float* emb_W  = (const float*)d_in[3];
    const float* emb_b  = (const float*)d_in[4];
    const float* conv_W = (const float*)d_in[5];
    const float* conv_b = (const float*)d_in[6];
    const float* bn_g   = (const float*)d_in[7];
    const float* bn_b   = (const float*)d_in[8];
    const float* W1 = (const float*)d_in[9];  const float* b1 = (const float*)d_in[10];
    const float* W2 = (const float*)d_in[11]; const float* b2 = (const float*)d_in[12];
    const float* W3 = (const float*)d_in[13]; const float* b3 = (const float*)d_in[14];
    float* out_final = (float*)d_out;

    const int* src = ei;
    const int* dst = ei + N_EDGES;

    // workspace carve-out
    char* w = (char*)d_ws;
    auto alloc = [&](size_t bytes) -> void* {
        void* p = (void*)w;
        w += (bytes + 255) & ~(size_t)255;
        return p;
    };
    float* deg   = (float*)alloc((size_t)N_NODES * 4);
    float* lsum  = (float*)alloc(4);
    float* amp   = (float*)alloc((size_t)N_NODES * 4);
    float* att   = (float*)alloc((size_t)N_NODES * 4);
    float* h     = (float*)alloc((size_t)N_NODES * DIM * 4);
    float* outb  = (float*)alloc((size_t)N_NODES * DIM * 4);
    float* agg4  = (float*)alloc((size_t)N_NODES * 512 * 4);
    bf16*  aggB  = (bf16*)alloc((size_t)N_NODES * 512 * 2);
    bf16*  Wbt   = (bf16*)alloc((size_t)LAYERS * DIM * 12 * DIM * 2);
    float* bnsum = (float*)alloc(DIM * 4);
    float* bnsq  = (float*)alloc(DIM * 4);
    float* gsum  = (float*)alloc((size_t)GRAPHS * DIM * 4);
    float* gcnt  = (float*)alloc((size_t)GRAPHS * 4);
    float* t1    = (float*)alloc((size_t)GRAPHS * 64 * 4);
    float* t2    = (float*)alloc((size_t)GRAPHS * 32 * 4);

    const int B = 256;

    // degree + PNA scalers
    k_fill_f32<<<gdiv(N_NODES, B), B, 0, stream>>>(deg, 0.0f, N_NODES);
    k_fill_f32<<<1, 32, 0, stream>>>(lsum, 0.0f, 1);
    k_deg<<<gdiv(N_EDGES, B), B, 0, stream>>>(dst, deg, N_EDGES);
    k_logsum<<<gdiv(N_NODES, B), B, 0, stream>>>(deg, lsum, N_NODES);
    k_ampatt<<<gdiv(N_NODES, B), B, 0, stream>>>(deg, lsum, amp, att, N_NODES);

    // embedding + weight transpose/convert
    k_embed<<<gdiv((long long)N_NODES * DIM, B), B, 0, stream>>>(x, emb_W, emb_b, h, N_NODES * DIM);
    k_convw<<<gdiv((long long)LAYERS * 12 * DIM * DIM, B), B, 0, stream>>>(
        conv_W, Wbt, LAYERS * 12 * DIM * DIM);

    const int ROWT   = N_NODES / 16;                 // 3125
    const int NBLK_G = 8 * ((ROWT + BROWT - 1) / BROWT);  // 8 col tiles x 782 row groups

    for (int l = 0; l < LAYERS; ++l) {
        k_init_agg<<<gdiv((long long)N_NODES * 512, B), B, 0, stream>>>(agg4, N_NODES * 512);
        k_fill_f32<<<1, 256, 0, stream>>>(bnsum, 0.0f, DIM);
        k_fill_f32<<<1, 256, 0, stream>>>(bnsq, 0.0f, DIM);

        k_scatter<<<gdiv((long long)N_EDGES * DIM, B), B, 0, stream>>>(
            src, dst, h, agg4, N_EDGES * DIM);
        k_fixup<<<gdiv((long long)N_NODES * DIM, B), B, 0, stream>>>(
            agg4, deg, aggB, N_NODES * DIM);

        k_gemm<<<NBLK_G, 128, 0, stream>>>(
            aggB, Wbt + (size_t)l * DIM * 12 * DIM, amp, att, conv_b + l * DIM, outb);

        k_bnstat<<<256, DIM, 0, stream>>>(outb, bnsum, bnsq, N_NODES);
        k_bnapply<<<gdiv((long long)N_NODES * DIM, B), B, 0, stream>>>(
            outb, bnsum, bnsq, bn_g + l * DIM, bn_b + l * DIM, h, N_NODES * DIM);
    }

    // pooling + MLP
    k_fill_f32<<<gdiv(GRAPHS * DIM, B), B, 0, stream>>>(gsum, 0.0f, GRAPHS * DIM);
    k_fill_f32<<<gdiv(GRAPHS, B), B, 0, stream>>>(gcnt, 0.0f, GRAPHS);
    k_pool<<<gdiv((long long)N_NODES * DIM, B), B, 0, stream>>>(h, batch, gsum, gcnt, N_NODES * DIM);
    k_gmean<<<gdiv(GRAPHS * DIM, B), B, 0, stream>>>(gsum, gcnt, GRAPHS * DIM);
    k_mlp1<<<gdiv(GRAPHS * 64, B), B, 0, stream>>>(gsum, W1, b1, t1, GRAPHS * 64);
    k_mlp2<<<gdiv(GRAPHS * 32, B), B, 0, stream>>>(t1, W2, b2, t2, GRAPHS * 32);
    k_mlp3<<<gdiv(GRAPHS * TOUT, B), B, 0, stream>>>(t2, W3, b3, out_final, GRAPHS * TOUT);
}
